// GraphSageNet_76785425318033
// MI455X (gfx1250) — compile-verified, matrix operand order checked by reading, not run on hardware
//
#include <hip/hip_runtime.h>
#include <hip/hip_bf16.h>

// ---------------------------------------------------------------------------
// GraphSAGE on MI455X (gfx1250, wave32, WMMA).
//
// Roofline: ~80 GFLOP total, dominated by the 800k-edge MLP (~66 GFLOP) and
// by irregular gather/scatter traffic. bf16 WMMA (fp32 accumulate) keeps the
// compute under the gather floor. This revision mirrors h in bf16 so the
// dominant edge gathers (4x aggregation + readout) move half the bytes
// (~2.5 GB -> ~1.3 GB at 23.3 TB/s) and A-fragments load directly as bf16
// with zero conversion VALU. fp32 h is kept for exact residuals; the mean
// aggregation accumulates in fp32 atomics.
// ---------------------------------------------------------------------------

typedef __attribute__((ext_vector_type(16))) __bf16 v16bf;
typedef __attribute__((ext_vector_type(8)))  __bf16 v8bf;
typedef __attribute__((ext_vector_type(8)))  float  v8f;

#define N_NODES   50000
#define N_EDGES   800000
#define IN_DIM    64
#define HID       128
#define OUTD      128
#define NCLS      2
#define NLAYERS   4

// f32 -> bf16: let the backend pick the native convert (gfx1250 has bf16 ALU)
__device__ __forceinline__ __bf16 f2bf(float f) { return (__bf16)f; }

__device__ __forceinline__ v8f wmma_bf16(v16bf a, v16bf b, v8f c) {
    // (neg_a, A, neg_b, B, c_mod, C, reuse_a, reuse_b)
    return __builtin_amdgcn_wmma_f32_16x16x32_bf16(false, a, false, b,
                                                   (short)0, c, false, false);
}

// A-fragment (16x32 bf16, ISA layout): lane holds row m = lane&15;
// half = lane>>4 selects K runs [kb..kb+7] and [kb+16..kb+23], kb = 8*half.
// fp32 source (converts), p = row base + 32*kk.
template <typename P>
__device__ __forceinline__ v16bf load_a_frag_f32(P p, int half) {
    v16bf a;
    p += half * 8;
#pragma unroll
    for (int i = 0; i < 8; ++i) a[i] = f2bf(p[i]);
#pragma unroll
    for (int i = 0; i < 8; ++i) a[8 + i] = f2bf(p[16 + i]);
    return a;
}

// bf16 source (no conversion): two 16B loads per lane.
template <typename P>
__device__ __forceinline__ v16bf load_a_frag_bf(P p, int half) {
    v8bf lo = *(const v8bf*)(p + half * 8);
    v8bf hi = *(const v8bf*)(p + half * 8 + 16);
    v16bf a;
#pragma unroll
    for (int i = 0; i < 8; ++i) { a[i] = lo[i]; a[8 + i] = hi[i]; }
    return a;
}

// ---------------------------------------------------------------------------
// Weight pre-swizzle: row-major fp32 W[K][N] -> fragment-major bf16 so a
// B-fragment is one contiguous 32B load per lane.
// ---------------------------------------------------------------------------
__global__ void pack_w_kernel(const float* __restrict__ W,
                              __bf16* __restrict__ dst, int K, int N) {
    int t = blockIdx.x * blockDim.x + threadIdx.x;
    int ktiles = K >> 5, ntiles = N >> 4;
    int total = ktiles * ntiles * 32;
    if (t >= total) return;
    int lane = t & 31;
    int nt   = (t >> 5) % ntiles;
    int kk   = (t >> 5) / ntiles;
    int half = lane >> 4;
    int n    = nt * 16 + (lane & 15);
    int kb   = kk * 32 + half * 8;
    __bf16* out = dst + (size_t)t * 16;
#pragma unroll
    for (int i = 0; i < 8; ++i) out[i]     = f2bf(W[(size_t)(kb + i) * N + n]);
#pragma unroll
    for (int i = 0; i < 8; ++i) out[8 + i] = f2bf(W[(size_t)(kb + 16 + i) * N + n]);
}

// ---------------------------------------------------------------------------
// Degree / inverse degree (edges are static -> computed once per launch)
// ---------------------------------------------------------------------------
__global__ void degree_kernel(const int* __restrict__ dst, int* __restrict__ deg) {
    int t = blockIdx.x * blockDim.x + threadIdx.x;
    if (t < N_EDGES) atomicAdd(&deg[dst[t]], 1);
}

__global__ void invdeg_kernel(const int* __restrict__ deg, float* __restrict__ inv) {
    int t = blockIdx.x * blockDim.x + threadIdx.x;
    if (t < N_NODES) inv[t] = 1.0f / fmaxf((float)deg[t], 1.0f);
}

// ---------------------------------------------------------------------------
// Embedding: h = x @ W_embed + b  (M=50000, K=64, N=128). One wave per
// 16x128 tile. Writes fp32 h and the bf16 mirror.
// ---------------------------------------------------------------------------
__global__ void __launch_bounds__(32)
embed_kernel(const float* __restrict__ x, const __bf16* __restrict__ wf,
             const float* __restrict__ bias, float* __restrict__ h,
             __bf16* __restrict__ hb) {
    const int lane = threadIdx.x;
    const int m = lane & 15, half = lane >> 4;
    const int mtile = blockIdx.x;
    const float* arow = x + (size_t)(mtile * 16 + m) * IN_DIM;

    v8f zero = {};
    v8f acc[8];
#pragma unroll
    for (int nt = 0; nt < 8; ++nt) acc[nt] = zero;

#pragma unroll
    for (int kk = 0; kk < 2; ++kk) {
        v16bf a = load_a_frag_f32(arow + 32 * kk, half);
#pragma unroll
        for (int nt = 0; nt < 8; ++nt) {
            v16bf b = *(const v16bf*)(wf + ((size_t)(kk * 8 + nt) * 32 + lane) * 16);
            acc[nt] = wmma_bf16(a, b, acc[nt]);
        }
    }
    // C/D layout: lane holds col nt*16+m of rows (v + 8*half)
#pragma unroll
    for (int nt = 0; nt < 8; ++nt) {
        int col = nt * 16 + m;
        float bv = bias[col];
#pragma unroll
        for (int v = 0; v < 8; ++v) {
            int r = mtile * 16 + v + 8 * half;
            float o = acc[nt][v] + bv;
            h[(size_t)r * HID + col]  = o;
            hb[(size_t)r * HID + col] = f2bf(o);
        }
    }
}

// ---------------------------------------------------------------------------
// Mean-aggregation scatter: msum[dst] += h[src]. One wave per edge; gathers
// the bf16 mirror (half the bytes), accumulates with fp32 atomics.
// ---------------------------------------------------------------------------
__global__ void aggregate_kernel(const __bf16* __restrict__ hb,
                                 const int* __restrict__ src,
                                 const int* __restrict__ dstv,
                                 float* __restrict__ msum) {
    int wid  = (blockIdx.x * blockDim.x + threadIdx.x) >> 5;
    int lane = threadIdx.x & 31;
    if (wid >= N_EDGES) return;
    int s = src[wid], d = dstv[wid];
    const __bf16* hs = hb + (size_t)s * HID;
    float* ms = msum + (size_t)d * HID;
#pragma unroll
    for (int j = 0; j < 4; ++j)
        atomicAdd(&ms[lane + 32 * j], (float)hs[lane + 32 * j]);
}

// ---------------------------------------------------------------------------
// SAGE layer: bundle = concat(h, msum/deg) @ W + b ; per-row L2 normalize
// (shfl_xor over 16-lane halves, no LDS) ; h += relu(bundle/norm).
// One wave per 16-row tile; tile rows are exclusive -> in-place update safe.
// ---------------------------------------------------------------------------
__global__ void __launch_bounds__(32)
sage_kernel(float* __restrict__ h, __bf16* __restrict__ hb,
            const float* __restrict__ msum, const float* __restrict__ invdeg,
            const __bf16* __restrict__ wf, const float* __restrict__ bias) {
    const int lane = threadIdx.x;
    const int m = lane & 15, half = lane >> 4;
    const int mtile = blockIdx.x;
    const int row = mtile * 16 + m;
    const __bf16* hrow = hb + (size_t)row * HID;
    const float*  mrow = msum + (size_t)row * HID;
    const float   idg  = invdeg[row];

    v8f zero = {};
    v8f acc[8];
#pragma unroll
    for (int nt = 0; nt < 8; ++nt) acc[nt] = zero;

#pragma unroll
    for (int kk = 0; kk < 8; ++kk) {
        v16bf a;
        if (kk < 4) {
            a = load_a_frag_bf(hrow + 32 * kk, half);     // bf16, no convert
        } else {                                          // c = msum * invdeg
            const float* p = mrow + 32 * (kk - 4) + half * 8;
#pragma unroll
            for (int i = 0; i < 8; ++i) a[i]     = f2bf(p[i] * idg);
#pragma unroll
            for (int i = 0; i < 8; ++i) a[8 + i] = f2bf(p[16 + i] * idg);
        }
#pragma unroll
        for (int nt = 0; nt < 8; ++nt) {
            v16bf b = *(const v16bf*)(wf + ((size_t)(kk * 8 + nt) * 32 + lane) * 16);
            acc[nt] = wmma_bf16(a, b, acc[nt]);
        }
    }
#pragma unroll
    for (int nt = 0; nt < 8; ++nt) {
        float bv = bias[nt * 16 + m];
#pragma unroll
        for (int v = 0; v < 8; ++v) acc[nt][v] += bv;
    }
    // row L2 norms: in-lane partial over N-tiles, xor-reduce within the half
    float inv_n[8];
#pragma unroll
    for (int v = 0; v < 8; ++v) {
        float s = 0.f;
#pragma unroll
        for (int nt = 0; nt < 8; ++nt) s += acc[nt][v] * acc[nt][v];
#pragma unroll
        for (int mask = 1; mask <= 8; mask <<= 1) s += __shfl_xor(s, mask, 32);
        inv_n[v] = 1.0f / fmaxf(sqrtf(s), 1e-12f);
    }
    // residual: h += relu(bundle / norm); refresh bf16 mirror
#pragma unroll
    for (int nt = 0; nt < 8; ++nt) {
        int col = nt * 16 + m;
#pragma unroll
        for (int v = 0; v < 8; ++v) {
            int r = mtile * 16 + v + 8 * half;
            float o  = fmaxf(acc[nt][v] * inv_n[v], 0.f);
            float hv = h[(size_t)r * HID + col] + o;
            h[(size_t)r * HID + col]  = hv;
            hb[(size_t)r * HID + col] = f2bf(hv);
        }
    }
}

// ---------------------------------------------------------------------------
// Fused edge readout: e = concat(h[src], h[dst]) -> relu(e@W0+b0)
// -> relu(@W1+b1) -> @W2+b2. One wave per 16 edges. Inter-GEMM activations
// live in LDS as bf16 with padded strides (272B/144B rows: odd multiples of
// 16B, so the 16 row-readers spread across the 64 banks).
// ---------------------------------------------------------------------------
#define S1 136   // act1 row stride in bf16 (128 + 8 pad)
#define S2 72    // act2 row stride in bf16 (64 + 8 pad)

__global__ void __launch_bounds__(32)
readout_kernel(const __bf16* __restrict__ hb,
               const int* __restrict__ esrc, const int* __restrict__ edst,
               const __bf16* __restrict__ wf0, const float* __restrict__ b0,
               const __bf16* __restrict__ wf1, const float* __restrict__ b1,
               const float* __restrict__ W2,  const float* __restrict__ b2,
               float* __restrict__ out) {
    __shared__ __bf16 act1[16 * S1];
    __shared__ __bf16 act2[16 * S2];
    const int lane = threadIdx.x;
    const int m = lane & 15, half = lane >> 4;
    const int e = blockIdx.x * 16 + m;
    const __bf16* hs = hb + (size_t)esrc[e] * OUTD;
    const __bf16* hd = hb + (size_t)edst[e] * OUTD;

    v8f zero = {};
    // ---- GEMM1: [16x256] x [256x128] ----
    v8f acc[8];
#pragma unroll
    for (int nt = 0; nt < 8; ++nt) acc[nt] = zero;
#pragma unroll
    for (int kk = 0; kk < 8; ++kk) {
        const __bf16* base = (kk < 4) ? (hs + 32 * kk) : (hd + 32 * (kk - 4));
        v16bf a = load_a_frag_bf(base, half);
#pragma unroll
        for (int nt = 0; nt < 8; ++nt) {
            v16bf b = *(const v16bf*)(wf0 + ((size_t)(kk * 8 + nt) * 32 + lane) * 16);
            acc[nt] = wmma_bf16(a, b, acc[nt]);
        }
    }
#pragma unroll
    for (int nt = 0; nt < 8; ++nt) {
        int col = nt * 16 + m;
        float bv = b0[col];
#pragma unroll
        for (int v = 0; v < 8; ++v)
            act1[(v + 8 * half) * S1 + col] = f2bf(fmaxf(acc[nt][v] + bv, 0.f));
    }
    __syncthreads();

    // ---- GEMM2: [16x128] x [128x64] ----
    v8f acc2[4];
#pragma unroll
    for (int nt = 0; nt < 4; ++nt) acc2[nt] = zero;
#pragma unroll
    for (int kk = 0; kk < 4; ++kk) {
        v16bf a = load_a_frag_bf(&act1[m * S1 + 32 * kk], half);
#pragma unroll
        for (int nt = 0; nt < 4; ++nt) {
            v16bf b = *(const v16bf*)(wf1 + ((size_t)(kk * 4 + nt) * 32 + lane) * 16);
            acc2[nt] = wmma_bf16(a, b, acc2[nt]);
        }
    }
#pragma unroll
    for (int nt = 0; nt < 4; ++nt) {
        int col = nt * 16 + m;
        float bv = b1[col];
#pragma unroll
        for (int v = 0; v < 8; ++v)
            act2[(v + 8 * half) * S2 + col] = f2bf(fmaxf(acc2[nt][v] + bv, 0.f));
    }
    __syncthreads();

    // ---- tail: [16x64] x [64x2] scalar (lane: row = lane&15, class = lane>>4)
    {
        int r = lane & 15, c = lane >> 4;
        float s = b2[c];
#pragma unroll 8
        for (int k = 0; k < 64; ++k) s += (float)act2[r * S2 + k] * W2[k * NCLS + c];
        out[(size_t)(blockIdx.x * 16 + r) * NCLS + c] = s;
    }
}

// ---------------------------------------------------------------------------
extern "C" void kernel_launch(void* const* d_in, const int* in_sizes, int n_in,
                              void* d_out, int out_size, void* d_ws, size_t ws_size,
                              hipStream_t stream) {
    const float* x       = (const float*)d_in[0];
    const int*   esrc    = (const int*)  d_in[1];
    const int*   edst    = (const int*)  d_in[2];
    // d_in[3]=snorm_n, d_in[4]=snorm_e : unused by the reference
    const float* W_embed = (const float*)d_in[5];
    const float* b_embed = (const float*)d_in[6];
    const float* W_conv  = (const float*)d_in[7];   // [4][256][128]
    const float* b_conv  = (const float*)d_in[8];   // [4][128]
    const float* W_mlp0  = (const float*)d_in[9];
    const float* b_mlp0  = (const float*)d_in[10];
    const float* W_mlp1  = (const float*)d_in[11];
    const float* b_mlp1  = (const float*)d_in[12];
    const float* W_mlp2  = (const float*)d_in[13];
    const float* b_mlp2  = (const float*)d_in[14];

    // ---- carve workspace (256B aligned) ----
    char* ws = (char*)d_ws;
    size_t off = 0;
    auto carve = [&](size_t bytes) -> void* {
        void* p = ws + off;
        off = (off + bytes + 255) & ~(size_t)255;
        return p;
    };
    float*   h    = (float*)  carve((size_t)N_NODES * HID * 4);   // 25.6 MB
    float*   msum = (float*)  carve((size_t)N_NODES * HID * 4);   // 25.6 MB
    __bf16*  hbf  = (__bf16*) carve((size_t)N_NODES * HID * 2);   // 12.8 MB
    int*     deg  = (int*)    carve((size_t)N_NODES * 4);
    float*   invd = (float*)  carve((size_t)N_NODES * 4);
    const size_t FRAG_EMBED = (size_t)2 * 8 * 32 * 16;            // bf16 elems
    const size_t FRAG_CONV  = (size_t)8 * 8 * 32 * 16;
    const size_t FRAG_MLP0  = (size_t)8 * 8 * 32 * 16;
    const size_t FRAG_MLP1  = (size_t)4 * 4 * 32 * 16;
    __bf16* wfE = (__bf16*) carve(FRAG_EMBED * 2);
    __bf16* wfC = (__bf16*) carve(NLAYERS * FRAG_CONV * 2);
    __bf16* wf0 = (__bf16*) carve(FRAG_MLP0 * 2);
    __bf16* wf1 = (__bf16*) carve(FRAG_MLP1 * 2);

    // ---- weight swizzle + degrees (static per launch) ----
    hipMemsetAsync(deg, 0, (size_t)N_NODES * 4, stream);
    pack_w_kernel<<<(2 * 8 * 32 + 255) / 256, 256, 0, stream>>>(W_embed, wfE, IN_DIM, HID);
    for (int l = 0; l < NLAYERS; ++l)
        pack_w_kernel<<<(8 * 8 * 32 + 255) / 256, 256, 0, stream>>>(
            W_conv + (size_t)l * 256 * 128, wfC + (size_t)l * FRAG_CONV, 256, 128);
    pack_w_kernel<<<(8 * 8 * 32 + 255) / 256, 256, 0, stream>>>(W_mlp0, wf0, 256, 128);
    pack_w_kernel<<<(4 * 4 * 32 + 255) / 256, 256, 0, stream>>>(W_mlp1, wf1, 128, 64);
    degree_kernel<<<(N_EDGES + 255) / 256, 256, 0, stream>>>(edst, deg);
    invdeg_kernel<<<(N_NODES + 255) / 256, 256, 0, stream>>>(deg, invd);

    // ---- embedding ----
    embed_kernel<<<N_NODES / 16, 32, 0, stream>>>(x, wfE, b_embed, h, hbf);

    // ---- 4 SAGE layers ----
    for (int l = 0; l < NLAYERS; ++l) {
        hipMemsetAsync(msum, 0, (size_t)N_NODES * HID * 4, stream);
        aggregate_kernel<<<(N_EDGES * 32) / 256, 256, 0, stream>>>(hbf, esrc, edst, msum);
        sage_kernel<<<N_NODES / 16, 32, 0, stream>>>(
            h, hbf, msum, invd, wfC + (size_t)l * FRAG_CONV, b_conv + (size_t)l * HID);
    }

    // ---- fused edge MLP readout ----
    readout_kernel<<<N_EDGES / 16, 32, 0, stream>>>(
        hbf, esrc, edst, wf0, b_mlp0, wf1, b_mlp1, W_mlp2, b_mlp2, (float*)d_out);
}
